// Encoder_82300163326282
// MI455X (gfx1250) — matrix operand order repaired and checked by reading, not compiled, not run on hardware
//
#include <hip/hip_runtime.h>

#define D 256
#define TN 8   // 8 column tiles of 16 -> 128 cols per wave

typedef __attribute__((ext_vector_type(16))) __bf16 v16bf;
typedef __attribute__((ext_vector_type(8)))  float  v8f;
typedef int v4i __attribute__((vector_size(16)));
typedef __attribute__((address_space(1))) v4i* gv4i_p;   // global int4*
typedef __attribute__((address_space(3))) v4i* lv4i_p;   // LDS int4*

union B16x16 {
    struct { uint4 lo; uint4 hi; } u;
    v16bf v;
};

#if defined(__has_builtin)
#if __has_builtin(__builtin_amdgcn_global_load_async_to_lds_b128) && \
    __has_builtin(__builtin_amdgcn_s_wait_asynccnt)
#define USE_ASYNC_LDS 1
#endif
#endif
#ifndef USE_ASYNC_LDS
#define USE_ASYNC_LDS 0
#endif

// ---------------- utility kernels ----------------

__global__ void zero_f4(float4* __restrict__ p, long n4) {
    long t = (long)blockIdx.x * blockDim.x + threadIdx.x;
    if (t < n4) p[t] = make_float4(0.f, 0.f, 0.f, 0.f);
}

__global__ void copy_f4(const float4* __restrict__ src, float4* __restrict__ dst, long n4) {
    long t = (long)blockIdx.x * blockDim.x + threadIdx.x;
    if (t < n4) dst[t] = src[t];
}

__global__ void degree_kernel(const int* __restrict__ dst, float* __restrict__ cnt, int E) {
    int t = blockIdx.x * blockDim.x + threadIdx.x;
    if (t < E) atomicAdd(&cnt[dst[t]], 1.0f);
}

// agg[dst] += x[src], float4 chunks, 64 chunks per edge
__global__ void scatter_kernel(const int* __restrict__ src, const int* __restrict__ dst,
                               const float* __restrict__ x, float* __restrict__ agg, long total) {
    long t = (long)blockIdx.x * blockDim.x + threadIdx.x;
    if (t >= total) return;
    int e = (int)(t >> 6);
    int c = (int)(t & 63) << 2;
    int s = src[e], d = dst[e];
    float4 v = *(const float4*)(x + (size_t)s * D + c);
    float* ap = agg + (size_t)d * D + c;
    atomicAdd(ap + 0, v.x);
    atomicAdd(ap + 1, v.y);
    atomicAdd(ap + 2, v.z);
    atomicAdd(ap + 3, v.w);
}

// Wt[n][k] = bf16(W[k][n]), 256x256
__global__ void wtrans_kernel(const float* __restrict__ W, __bf16* __restrict__ Wt) {
    int t = blockIdx.x * blockDim.x + threadIdx.x;
    if (t >= D * D) return;
    int n = t >> 8, k = t & 255;
    Wt[t] = (__bf16)W[k * D + n];
}

// meanbf = bf16(agg / max(cnt,1)), hbf = bf16(hin)
__global__ void meanconv_kernel(const float* __restrict__ agg, const float* __restrict__ cnt,
                                const float* __restrict__ hin,
                                __bf16* __restrict__ meanbf, __bf16* __restrict__ hbf, long total) {
    long t = (long)blockIdx.x * blockDim.x + threadIdx.x;
    if (t >= total) return;
    int row = (int)(t >> 8);
    float inv = 1.0f / fmaxf(cnt[row], 1.0f);
    meanbf[t] = (__bf16)(agg[t] * inv);
    hbf[t]    = (__bf16)hin[t];
}

// ---------------- WMMA tile loaders ----------------
// A tile 16x32 bf16, row-major source with row stride 256.
// lanes 0-15 : row r, elems 0..7 = K k0..k0+7,    elems 8..15 = K k0+16..k0+23
// lanes 16-31: row r, elems 0..7 = K k0+8..k0+15, elems 8..15 = K k0+24..k0+31
__device__ __forceinline__ v16bf load_a_tile(const __bf16* __restrict__ base,
                                             int row0, int k0, int lane) {
    int half = lane >> 4;
    int r = lane & 15;
    const __bf16* p = base + (size_t)(row0 + r) * D + k0 + half * 8;
    B16x16 t;
    t.u.lo = *(const uint4*)(p);
    t.u.hi = *(const uint4*)(p + 16);
    return t.v;
}

// B tile 32x16 bf16 from LDS slab laid out [256 n][32 k].
// lanes 0-15 : col n, K = 0..7,16..23 of slab ; lanes 16-31: K = 8..15,24..31
__device__ __forceinline__ v16bf load_b_lds(const __bf16* slab, int n0, int lane) {
    int half = lane >> 4;
    int c = lane & 15;
    const __bf16* p = slab + (n0 + c) * 32 + half * 8;
    B16x16 t;
    t.u.lo = *(const uint4*)(p);
    t.u.hi = *(const uint4*)(p + 16);
    return t.v;
}

// copy one 64-byte slab row global -> LDS
__device__ __forceinline__ void stage_row64(const __bf16* g, __bf16* l) {
#if USE_ASYNC_LDS
#pragma unroll
    for (int i = 0; i < 4; ++i)
        __builtin_amdgcn_global_load_async_to_lds_b128(
            (gv4i_p)(g + i * 8), (lv4i_p)(l + i * 8), 0, 0);
#else
    uint4 t0 = ((const uint4*)g)[0];
    uint4 t1 = ((const uint4*)g)[1];
    uint4 t2 = ((const uint4*)g)[2];
    uint4 t3 = ((const uint4*)g)[3];
    ((uint4*)l)[0] = t0;
    ((uint4*)l)[1] = t1;
    ((uint4*)l)[2] = t2;
    ((uint4*)l)[3] = t3;
#endif
}

__device__ __forceinline__ void wait_stage() {
#if USE_ASYNC_LDS
    __builtin_amdgcn_s_wait_asynccnt(0);
#endif
}

// ---------------- fused SAGE GEMM: out = leaky(mean@Wl + x@Wr + b) ----------------
// Block: 256 threads = 8 waves; block tile = 64 rows x 256 cols.
// Weight slabs (both matrices, 32 K x 256 N, bf16) double-buffered in LDS.
__global__ __launch_bounds__(256)
void sage_gemm_kernel(const __bf16* __restrict__ meanbf, const __bf16* __restrict__ xbf,
                      const __bf16* __restrict__ Wlt, const __bf16* __restrict__ Wrt,
                      const float* __restrict__ bias, float* __restrict__ out, int Nrows) {
    __shared__ __bf16 sW[2][2][D * 32];   // [buf][mat][256n x 32k] = 64 KB

    const int tid  = threadIdx.x;
    const int wave = tid >> 5;
    const int lane = tid & 31;

    int row0 = blockIdx.x * 64 + (wave >> 1) * 16;
    if (row0 > Nrows - 16) row0 = Nrows - 16;   // tail: duplicate waves recompute identical rows
    const int n0base = (wave & 1) * 128;

    // stage k-slab 0 into buffer 0 (thread tid copies slab row tid of each matrix)
    stage_row64(Wlt + (size_t)tid * D + 0, &sW[0][0][tid * 32]);
    stage_row64(Wrt + (size_t)tid * D + 0, &sW[0][1][tid * 32]);

    // prefetch A fragments for k-step 0
    v16bf aM = load_a_tile(meanbf, row0, 0, lane);
    v16bf aX = load_a_tile(xbf,    row0, 0, lane);

    v8f acc[TN] = {};

    for (int kb = 0; kb < D / 32; ++kb) {
        const int buf = kb & 1;
        wait_stage();          // own async slab copies for buf done
        __syncthreads();       // all waves' copies visible; also guards buffer reuse

        if (kb < D / 32 - 1) {
            const int k1 = (kb + 1) * 32;
            stage_row64(Wlt + (size_t)tid * D + k1, &sW[buf ^ 1][0][tid * 32]);
            stage_row64(Wrt + (size_t)tid * D + k1, &sW[buf ^ 1][1][tid * 32]);
        }

        v16bf aMn = aM, aXn = aX;
        if (kb < D / 32 - 1) {
            const int k1 = (kb + 1) * 32;
            aMn = load_a_tile(meanbf, row0, k1, lane);
            aXn = load_a_tile(xbf,    row0, k1, lane);
        }

        const __bf16* slabL = &sW[buf][0][0];
        const __bf16* slabR = &sW[buf][1][0];
#pragma unroll
        for (int t = 0; t < TN; ++t) {
            const int n0 = n0base + t * 16;
            v16bf bl = load_b_lds(slabL, n0, lane);
            v16bf br = load_b_lds(slabR, n0, lane);
            acc[t] = __builtin_amdgcn_wmma_f32_16x16x32_bf16(
                false, aM, false, bl, (short)0, acc[t], false, false);
            acc[t] = __builtin_amdgcn_wmma_f32_16x16x32_bf16(
                false, aX, false, br, (short)0, acc[t], false, false);
        }
        aM = aMn;
        aX = aXn;
    }

    // epilogue: bias + LeakyReLU(0.5); C/D layout: VGPR v -> row row0+v+half*8, lane&15 -> col
    const int half = lane >> 4;
    const int r = lane & 15;
#pragma unroll
    for (int t = 0; t < TN; ++t) {
        const int col = n0base + t * 16 + r;
        const float bv = bias[col];
        float* op = out + (size_t)(row0 + half * 8) * D + col;
#pragma unroll
        for (int v = 0; v < 8; ++v) {
            float h = acc[t][v] + bv;
            h = (h >= 0.f) ? h : 0.5f * h;
            op[(size_t)v * D] = h;
        }
    }
}

// ---------------- host launcher ----------------

extern "C" void kernel_launch(void* const* d_in, const int* in_sizes, int n_in,
                              void* d_out, int out_size, void* d_ws, size_t ws_size,
                              hipStream_t stream) {
    const float* x   = (const float*)d_in[0];
    const int*   src = (const int*)d_in[1];
    const float* Wl0 = (const float*)d_in[2];
    const float* bl0 = (const float*)d_in[3];
    const float* Wr0 = (const float*)d_in[4];
    const float* Wl1 = (const float*)d_in[5];
    const float* bl1 = (const float*)d_in[6];
    const float* Wr1 = (const float*)d_in[7];

    const int N = in_sizes[0] / D;          // 50000
    const int E = in_sizes[1] / 2;          // 800000
    const int* dst = src + E;

    float* h_out = (float*)d_out;                     // first N*D: final h (layer-0 scratch)
    float* t_out = (float*)d_out + (size_t)N * D;     // second N*D: target = x

    // workspace layout
    char* ws = (char*)d_ws;
    float*  cnt    = (float*)(ws);                                   // N f32
    float*  agg    = (float*)(ws + (size_t)N * 4);                   // N*D f32 (contiguous w/ cnt)
    __bf16* meanbf = (__bf16*)(ws + (size_t)N * 4 + (size_t)N * D * 4);
    __bf16* xbf    = (__bf16*)((char*)meanbf + (size_t)N * D * 2);
    __bf16* wbase  = (__bf16*)((char*)xbf + (size_t)N * D * 2);
    __bf16* Wl0t = wbase;
    __bf16* Wr0t = wbase + 1 * D * D;
    __bf16* Wl1t = wbase + 2 * D * D;
    __bf16* Wr1t = wbase + 3 * D * D;

    const long ND = (long)N * D;
    const int  nRowTiles = N / 16;
    const int  THREADS = 256;

    auto blocks = [](long n, int t) { return (unsigned)((n + t - 1) / t); };

    // zero cnt + agg (contiguous)
    long z4 = (N + ND) / 4;
    zero_f4<<<blocks(z4, THREADS), THREADS, 0, stream>>>((float4*)cnt, z4);

    // degree counts (graph fixed across both layers)
    degree_kernel<<<blocks(E, THREADS), THREADS, 0, stream>>>(dst, cnt, E);

    // weight transposes to bf16
    wtrans_kernel<<<blocks(D * D, THREADS), THREADS, 0, stream>>>(Wl0, Wl0t);
    wtrans_kernel<<<blocks(D * D, THREADS), THREADS, 0, stream>>>(Wr0, Wr0t);
    wtrans_kernel<<<blocks(D * D, THREADS), THREADS, 0, stream>>>(Wl1, Wl1t);
    wtrans_kernel<<<blocks(D * D, THREADS), THREADS, 0, stream>>>(Wr1, Wr1t);

    const unsigned gemmBlocks = (unsigned)((nRowTiles + 3) / 4);   // 64 rows per block

    // ---- layer 0 ----
    long stotal = (long)E * 64;
    scatter_kernel<<<blocks(stotal, THREADS), THREADS, 0, stream>>>(src, dst, x, agg, stotal);
    meanconv_kernel<<<blocks(ND, THREADS), THREADS, 0, stream>>>(agg, cnt, x, meanbf, xbf, ND);
    sage_gemm_kernel<<<gemmBlocks, THREADS, 0, stream>>>(meanbf, xbf, Wl0t, Wr0t, bl0, h_out, N);

    // ---- layer 1 ----
    long a4 = ND / 4;
    zero_f4<<<blocks(a4, THREADS), THREADS, 0, stream>>>((float4*)agg, a4);
    scatter_kernel<<<blocks(stotal, THREADS), THREADS, 0, stream>>>(src, dst, h_out, agg, stotal);
    meanconv_kernel<<<blocks(ND, THREADS), THREADS, 0, stream>>>(agg, cnt, h_out, meanbf, xbf, ND);
    sage_gemm_kernel<<<gemmBlocks, THREADS, 0, stream>>>(meanbf, xbf, Wl1t, Wr1t, bl1, h_out, N);

    // target = x
    copy_f4<<<blocks(ND / 4, THREADS), THREADS, 0, stream>>>((const float4*)x, (float4*)t_out, ND / 4);
}
